// GRUNetwork_22308060136257
// MI455X (gfx1250) — matrix operand order repaired and checked by reading, not compiled, not run on hardware
//
#include <hip/hip_runtime.h>
#include <hip/hip_bf16.h>
#include <math.h>

// Problem constants (from the reference)
#define B_   256
#define T_   512
#define I_   28
#define H_   128
#define L_   6
#define O_   10
#define G3H  384                 // 3*H
#define M_   (B_ * T_)           // 131072 rows for the input-projection GEMM

// ---- WMMA fragment types (CDNA5 / gfx1250, wave32) ----
typedef __attribute__((ext_vector_type(16))) __bf16 bf16x16;
typedef __attribute__((ext_vector_type(8)))  __bf16 bf16x8;
typedef __attribute__((ext_vector_type(8)))  float  f32x8;

union AFrag { bf16x16 v; __bf16 e[16]; };   // 16x32 bf16 A or 32x16 bf16 B tile
union CFrag { f32x8   v; float  e[8];  };   // 16x16 f32 C/D tile

// Fast activations: single v_exp_f32 + raw v_rcp_f32 (no div_scale/Newton chain).
__device__ __forceinline__ float fast_sigmoid(float x) {
    return __builtin_amdgcn_rcpf(1.f + __expf(-x));
}
__device__ __forceinline__ float fast_tanh(float x) {
    // tanh(x) = 2/(1+e^-2x) - 1 ; exact limits at +-inf
    return __builtin_amdgcn_rcpf(1.f + __expf(-2.f * x)) * 2.f - 1.f;
}

// -------------------------------------------------------------------------
// gx = A[M,K] * W[N=384,K]^T + bias  (bf16 WMMA, f32 accumulate)
// One 16x16 output tile per wave; 8 waves per block.
// A-fragment layout (ISA 7.12.2, 16-bit A 16x32):
//   lane = half*16 + row;  element h -> K = (h<8?0:16) + half*8 + (h&7)
// B-fragment layout (16-bit B 32x16):
//   lanes 0-15: N=lane, K=0..7,16..23 ; lanes 16-31: N=lane-16, K=8..15,24..31
// C/D layout: VGPR r -> M = r + half*8, N = lane&15
// -------------------------------------------------------------------------
template <int K>
__global__ __launch_bounds__(256) void gemm_gx(const float* __restrict__ A,
                                               const float* __restrict__ W,
                                               const float* __restrict__ bias,
                                               float* __restrict__ out) {
    const int wave = threadIdx.x >> 5;
    const int lane = threadIdx.x & 31;
    const int row  = lane & 15;
    const int half = lane >> 4;

    const int tile = blockIdx.x * 8 + wave;   // global tile id
    const int mt   = tile / 24;               // 24 N-tiles (384/16)
    const int nt   = tile - mt * 24;
    const int m0   = mt * 16;
    const int n0   = nt * 16;

    const float* Arow = A + (size_t)(m0 + row) * K;  // A row for this lane
    const float* Wrow = W + (size_t)(n0 + row) * K;  // W row = B column

    CFrag c{};  // zero accumulator

    constexpr int KSTEPS = (K + 31) / 32;
#pragma unroll
    for (int ks = 0; ks < KSTEPS; ++ks) {
        const int k0 = ks * 32;
        AFrag a{}, b{};
#pragma unroll
        for (int h = 0; h < 16; ++h) {
            const int ka = k0 + ((h < 8) ? 0 : 16) + half * 8 + (h & 7);
            a.e[h] = (ka < K) ? (__bf16)Arow[ka] : (__bf16)0.f;
            const int kb = k0 + ((h < 8) ? 0 : 16) + half * 8 + (h & 7);
            b.e[h] = (kb < K) ? (__bf16)Wrow[kb] : (__bf16)0.f;
        }
        c.v = __builtin_amdgcn_wmma_f32_16x16x32_bf16(
                  false, a.v, false, b.v, (short)0, c.v, false, false);
    }

    const int   nCol = n0 + row;
    const float bv   = bias[nCol];
#pragma unroll
    for (int r = 0; r < 8; ++r) {
        const int m = m0 + r + half * 8;
        out[(size_t)m * G3H + nCol] = c.e[r] + bv;
    }
}

// -------------------------------------------------------------------------
// Recurrent scan for one layer. One block per 16-row batch tile (16 blocks).
// 8 waves; each wave owns 3 of the 24 gate-column tiles.
// W_hh is register-resident (12 bf16 B-fragments / wave) for the whole scan;
// h is LDS-resident (bf16 for WMMA A-frags + f32 master copy).
// gx is double-buffered in LDS via global_load_async_to_lds_b128: step t+1 is
// prefetched while step t's WMMAs run; s_wait_asynccnt syncs before the
// workgroup barrier.
// -------------------------------------------------------------------------
__global__ __launch_bounds__(256) void gru_scan(const float* __restrict__ gx,   // [B,T,384]
                                                const float* __restrict__ Whh,  // [384,128]
                                                const float* __restrict__ bhh,  // [384]
                                                float* __restrict__ hout) {     // [B,T,128]
    __shared__ __bf16 h_bf [16][H_];        // bf16 h for WMMA A fragments
    __shared__ float  h_f32[16][H_];        // exact f32 h for the z-blend
    __shared__ float  gh   [16][G3H];       // per-step h @ W_hh^T
    __shared__ float  gxbuf[2][16][G3H];    // double-buffered input projections
    __shared__ float  bsh  [G3H];           // b_hh

    const int tid   = threadIdx.x;
    const int wave  = tid >> 5;
    const int lane  = tid & 31;
    const int row   = lane & 15;
    const int half  = lane >> 4;
    const int bbase = blockIdx.x * 16;

    for (int i = tid; i < 16 * H_; i += 256) {
        (&h_bf[0][0])[i]  = (__bf16)0.f;
        (&h_f32[0][0])[i] = 0.f;
    }
    for (int i = tid; i < G3H; i += 256) bsh[i] = bhh[i];

    // Register-resident W_hh fragments: wave owns nt = wave + 8*i
    AFrag Bf[3][4];
#pragma unroll
    for (int i = 0; i < 3; ++i) {
        const int nt = wave + 8 * i;
        const float* Wr = Whh + (size_t)(nt * 16 + row) * H_;
#pragma unroll
        for (int ks = 0; ks < 4; ++ks) {
#pragma unroll
            for (int h = 0; h < 16; ++h) {
                const int k = ks * 32 + ((h < 8) ? 0 : 16) + half * 8 + (h & 7);
                Bf[i][ks].e[h] = (__bf16)Wr[k];
            }
        }
    }

    // ---- async gx prefetch setup: 16 rows x 1536B = 96 b128 chunks/row,
    // 1536 chunks total, 6 per thread.
    unsigned      lds_off[6];
    unsigned long gaddr[6];
#pragma unroll
    for (int i = 0; i < 6; ++i) {
        const int c  = tid + i * 256;        // 0..1535
        const int bl = c / 96;               // row in batch tile
        const int rb = (c - bl * 96) * 16;   // byte offset within 1536B row
        lds_off[i] = (unsigned)(bl * (G3H * 4) + rb);
        gaddr[i]   = (unsigned long)(const char*)gx +
                     (unsigned long)((size_t)(bbase + bl) * T_) * (G3H * 4) + rb;
    }
    const unsigned gxb0 = (unsigned)(unsigned long)(&gxbuf[0][0][0]);
    const unsigned gxb1 = (unsigned)(unsigned long)(&gxbuf[1][0][0]);

    // Prologue: issue async loads for t = 0 into buffer 0.
#pragma unroll
    for (int i = 0; i < 6; ++i) {
        asm volatile("global_load_async_to_lds_b128 %0, %1, off"
                     :: "v"(gxb0 + lds_off[i]), "v"(gaddr[i]) : "memory");
        gaddr[i] += G3H * 4;                 // advance to t = 1
    }
    __syncthreads();

    for (int t = 0; t < T_; ++t) {
        const int par = t & 1;

        // Prefetch step t+1 while this step's WMMAs run.
        if (t + 1 < T_) {
            const unsigned dst = par ? gxb0 : gxb1;   // buffer (t+1)&1
#pragma unroll
            for (int i = 0; i < 6; ++i) {
                asm volatile("global_load_async_to_lds_b128 %0, %1, off"
                             :: "v"(dst + lds_off[i]), "v"(gaddr[i]) : "memory");
                gaddr[i] += G3H * 4;
            }
        }

        // A fragments from LDS h (two 16B ds loads per k-step)
        AFrag a[4];
#pragma unroll
        for (int ks = 0; ks < 4; ++ks) {
            const int base = ks * 32 + half * 8;
            *(bf16x8*)&a[ks].e[0] = *(const bf16x8*)&h_bf[row][base];
            *(bf16x8*)&a[ks].e[8] = *(const bf16x8*)&h_bf[row][base + 16];
        }
#pragma unroll
        for (int i = 0; i < 3; ++i) {
            CFrag acc{};
#pragma unroll
            for (int ks = 0; ks < 4; ++ks)
                acc.v = __builtin_amdgcn_wmma_f32_16x16x32_bf16(
                            false, a[ks].v, false, Bf[i][ks].v,
                            (short)0, acc.v, false, false);
            const int n0 = (wave + 8 * i) * 16;
#pragma unroll
            for (int r = 0; r < 8; ++r)
                gh[r + half * 8][n0 + row] = acc.e[r];
        }

        // Ensure this step's 6 async chunks have landed (6 newer ones may be
        // in flight for t+1); then barrier so every wave's chunks are visible.
        if (t + 1 < T_) {
            asm volatile("s_wait_asynccnt 0x6" ::: "memory");
        } else {
            asm volatile("s_wait_asynccnt 0x0" ::: "memory");
        }
        __syncthreads();

        // Fused gate math: 16*128 = 2048 elements, 8 per thread
        const float* gxc = &gxbuf[par][0][0];
#pragma unroll
        for (int ii = 0; ii < 8; ++ii) {
            const int idx = tid + ii * 256;
            const int bl  = idx >> 7;
            const int j   = idx & 127;
            const float xr = gxc[bl * G3H + j];
            const float xz = gxc[bl * G3H + j + H_];
            const float xn = gxc[bl * G3H + j + 2 * H_];
            const float hr = gh[bl][j]          + bsh[j];
            const float hz = gh[bl][j + H_]     + bsh[j + H_];
            const float hn = gh[bl][j + 2 * H_] + bsh[j + 2 * H_];
            const float r_ = fast_sigmoid(xr + hr);
            const float z_ = fast_sigmoid(xz + hz);
            const float n_ = fast_tanh(xn + r_ * hn);
            const float hv = (1.f - z_) * n_ + z_ * h_f32[bl][j];
            h_f32[bl][j] = hv;
            h_bf[bl][j]  = (__bf16)hv;
            hout[((size_t)(bbase + bl) * T_ + t) * H_ + j] = hv;
        }
        __syncthreads();
    }
}

// -------------------------------------------------------------------------
// logits = h_last @ fc_w^T + fc_b ; log_softmax.  One wave32 per batch row.
// -------------------------------------------------------------------------
__global__ __launch_bounds__(32) void fc_logsoftmax(const float* __restrict__ hseq, // [B,T,H]
                                                    const float* __restrict__ fw,   // [10,128]
                                                    const float* __restrict__ fb,   // [10]
                                                    float* __restrict__ out) {      // [B,10]
    const int b    = blockIdx.x;
    const int lane = threadIdx.x;
    const float* hp = hseq + ((size_t)b * T_ + (T_ - 1)) * H_;

    float lg[O_];
#pragma unroll
    for (int o = 0; o < O_; ++o) {
        float s = 0.f;
#pragma unroll
        for (int j = lane; j < H_; j += 32) s += hp[j] * fw[o * H_ + j];
#pragma unroll
        for (int d = 16; d > 0; d >>= 1) s += __shfl_xor(s, d, 32);
        lg[o] = s + fb[o];
    }
    float m = lg[0];
#pragma unroll
    for (int o = 1; o < O_; ++o) m = fmaxf(m, lg[o]);
    float se = 0.f;
#pragma unroll
    for (int o = 0; o < O_; ++o) se += __expf(lg[o] - m);
    const float lse = m + __logf(se);
    if (lane < O_) out[b * O_ + lane] = lg[lane] - lse;
}

// -------------------------------------------------------------------------
extern "C" void kernel_launch(void* const* d_in, const int* in_sizes, int n_in,
                              void* d_out, int out_size, void* d_ws, size_t ws_size,
                              hipStream_t stream) {
    const float* x         = (const float*)d_in[0];  // [B,T,28]
    const float* W_ih0     = (const float*)d_in[1];  // [384,28]
    const float* W_ih_rest = (const float*)d_in[2];  // [5,384,128]
    const float* W_hh      = (const float*)d_in[3];  // [6,384,128]
    const float* b_ih      = (const float*)d_in[4];  // [6,384]
    const float* b_hh      = (const float*)d_in[5];  // [6,384]
    const float* fc_w      = (const float*)d_in[6];  // [10,128]
    const float* fc_b      = (const float*)d_in[7];  // [10]
    float* out = (float*)d_out;

    // Workspace: gx [M,384] f32 (192 MiB) + ping/pong h [M,128] f32 (64 MiB each)
    float* gx = (float*)d_ws;
    float* hA = gx + (size_t)M_ * G3H;
    float* hB = hA + (size_t)M_ * H_;

    const int gemm_blocks = (M_ / 16) * 24 / 8;  // 24576 blocks, 8 tiles each

    // Layer 0 (K = 28)
    gemm_gx<I_><<<gemm_blocks, 256, 0, stream>>>(x, W_ih0, b_ih, gx);
    gru_scan<<<16, 256, 0, stream>>>(gx, W_hh, b_hh, hA);

    const float* hin = hA;
    float*       ho  = hB;
    for (int l = 1; l < L_; ++l) {               // layers 1..5 (K = 128)
        gemm_gx<H_><<<gemm_blocks, 256, 0, stream>>>(
            hin, W_ih_rest + (size_t)(l - 1) * G3H * H_, b_ih + l * G3H, gx);
        gru_scan<<<16, 256, 0, stream>>>(
            gx, W_hh + (size_t)l * G3H * H_, b_hh + l * G3H, ho);
        float* tmp = (float*)hin; hin = ho; ho = tmp;
    }

    fc_logsoftmax<<<B_, 32, 0, stream>>>(hin, fc_w, fc_b, out);
}
